// RNNBinaryClassifier_74062416052987
// MI455X (gfx1250) — compile-verified, matrix operand order checked by reading, not run on hardware
//
#include <hip/hip_runtime.h>
#include <hip/hip_bf16.h>
#include <math.h>

// Problem constants (from reference): x[64,512,2048] fp32, H=128.
#define INPUT_DIM 2048
#define HIDDEN    128
#define BATCH     64
#define SEQ       512
#define K_STAGE   256            // K-columns of W staged in LDS per block (64 KB)

typedef __bf16 bf16x16 __attribute__((ext_vector_type(16)));
typedef __bf16 bf16x8  __attribute__((ext_vector_type(8)));
typedef float  f32x8   __attribute__((ext_vector_type(8)));
typedef int    v4i     __attribute__((vector_size(16)));

// 16-element bf16 WMMA operand assembled from two 16-byte halves
// (K = base..base+7 in lo, K = base+16..base+23 in hi, per the ISA A/B layout).
union BF16Vec {
    bf16x16 v;
    struct { bf16x8 lo, hi; } h;
};

// --- async global->LDS copy (CDNA5 GLOBAL_LOAD_ASYNC_TO_LDS_B128), guarded ---
#if defined(__has_builtin)
#if __has_builtin(__builtin_amdgcn_global_load_async_to_lds_b128)
#define HAVE_ASYNC_LDS 1
#endif
#endif

#ifdef HAVE_ASYNC_LDS
typedef __attribute__((address_space(1))) v4i* as1_v4i;   // global v4i*
typedef __attribute__((address_space(3))) v4i* as3_v4i;   // LDS v4i*
#endif

__device__ __forceinline__ void copy16_g2lds(const __bf16* src, __bf16* dst) {
#ifdef HAVE_ASYNC_LDS
    // flat low 32 bits are the LDS address (ISA 10.2)
    __builtin_amdgcn_global_load_async_to_lds_b128(
        (as1_v4i)(unsigned long long)src,
        (as3_v4i)(unsigned int)(unsigned long long)dst,
        0, 0);
#else
    *reinterpret_cast<bf16x8*>(dst) = *reinterpret_cast<const bf16x8*>(src);
#endif
}

__device__ __forceinline__ void wait_async_copies() {
#ifdef HAVE_ASYNC_LDS
#if __has_builtin(__builtin_amdgcn_s_wait_asynccnt)
    __builtin_amdgcn_s_wait_asynccnt(0);
#else
    asm volatile("s_wait_asynccnt 0" ::: "memory");
#endif
#endif
}

// ---------------------------------------------------------------------------
// Kernel 0: one-time W_ih fp32 -> bf16 conversion into workspace (512 KB,
// stays L2-resident for the whole GEMM). Native (__bf16) cast => RNE cvt op.
// ---------------------------------------------------------------------------
__global__ void cvt_to_bf16(const float* __restrict__ in,
                            __bf16* __restrict__ out, int n) {
    int i = blockIdx.x * blockDim.x + threadIdx.x;
    int stride = gridDim.x * blockDim.x;
    for (; i < n; i += stride) out[i] = (__bf16)in[i];
}

// ---------------------------------------------------------------------------
// Kernel 1: xw[t][b][h] = x[b,t,:] . W_ih[h,:] + b_ih[h] + b_hh[h]
// M = B*T = 32768 rows, N = 128, K = 2048, v_wmma_f32_16x16x32_bf16.
// One wave per 16-row M-tile, all 8 N-tiles per wave.
//  - x is streamed fp32 (mandatory 256 MB HBM traffic), converted to bf16 in
//    registers, with next-chunk loads hoisted over the WMMA burst.
//  - W (bf16) staged per 256-K slab into LDS via async global->LDS copies:
//    all 8 waves share one copy (8x less L2 B-traffic).
//  - B operands ping-pong through two register buffers so each ds_load pair
//    overlaps the previous tile's WMMA instead of stalling on dscnt==0.
// Output stored time-major so the recurrence reads contiguous slabs.
// ---------------------------------------------------------------------------
__global__ __launch_bounds__(256) void xw_gemm(
        const float* __restrict__ x,
        const __bf16* __restrict__ Wbf,              // [HIDDEN][INPUT_DIM] bf16
        const float* __restrict__ b_ih,
        const float* __restrict__ b_hh,
        float* __restrict__ xw)                      // [SEQ][BATCH][HIDDEN]
{
    extern __shared__ __bf16 sB[];                   // [HIDDEN][K_STAGE] = 64 KB

    const int tid  = threadIdx.x;
    const int wave = tid >> 5;
    const int lane = tid & 31;
    const int hlf  = lane >> 4;        // which 16-lane half
    const int l    = lane & 15;
    const int mtile = blockIdx.x * 8 + wave;

    const float* xrow = x + (size_t)(mtile * 16 + l) * INPUT_DIM;  // A row M = l

    // Bias depends only on N -> broadcast across the 8 accumulator VGPRs.
    f32x8 acc[8];
#pragma unroll
    for (int nt = 0; nt < 8; ++nt) {
        float bb = b_ih[nt * 16 + l] + b_hh[nt * 16 + l];
#pragma unroll
        for (int r = 0; r < 8; ++r) acc[nt][r] = bb;
    }

    for (int ks = 0; ks < INPUT_DIM; ks += K_STAGE) {
        // ---- cooperative stage of W[:, ks..ks+K_STAGE) into LDS -----------
        // 128 rows x 256 cols bf16 = 4096 16-byte vectors; 16 per thread.
        for (int i = tid; i < (HIDDEN * K_STAGE) / 8; i += 256) {
            const int row = i >> 5;                  // K_STAGE/8 = 32 vec/row
            const int c8  = (i & 31) * 8;
            copy16_g2lds(Wbf + (size_t)row * INPUT_DIM + ks + c8,
                         &sB[row * K_STAGE + c8]);
        }
        wait_async_copies();
        __syncthreads();

        // ---- compute over this K slab -------------------------------------
        // Pipeline the x stream: raw fp32 chunk preloaded one step ahead.
        int kg = ks + hlf * 8;
        float4 r0 = *reinterpret_cast<const float4*>(xrow + kg);
        float4 r1 = *reinterpret_cast<const float4*>(xrow + kg + 4);
        float4 r2 = *reinterpret_cast<const float4*>(xrow + kg + 16);
        float4 r3 = *reinterpret_cast<const float4*>(xrow + kg + 20);

#pragma unroll
        for (int kc = 0; kc < K_STAGE; kc += 32) {
            BF16Vec a;
            a.v[0]  = (__bf16)r0.x; a.v[1]  = (__bf16)r0.y;
            a.v[2]  = (__bf16)r0.z; a.v[3]  = (__bf16)r0.w;
            a.v[4]  = (__bf16)r1.x; a.v[5]  = (__bf16)r1.y;
            a.v[6]  = (__bf16)r1.z; a.v[7]  = (__bf16)r1.w;
            a.v[8]  = (__bf16)r2.x; a.v[9]  = (__bf16)r2.y;
            a.v[10] = (__bf16)r2.z; a.v[11] = (__bf16)r2.w;
            a.v[12] = (__bf16)r3.x; a.v[13] = (__bf16)r3.y;
            a.v[14] = (__bf16)r3.z; a.v[15] = (__bf16)r3.w;

            if (kc + 32 < K_STAGE) {                 // next chunk's x loads
                const int kn = ks + kc + 32 + hlf * 8;
                r0 = *reinterpret_cast<const float4*>(xrow + kn);
                r1 = *reinterpret_cast<const float4*>(xrow + kn + 4);
                r2 = *reinterpret_cast<const float4*>(xrow + kn + 16);
                r3 = *reinterpret_cast<const float4*>(xrow + kn + 20);
                __builtin_prefetch(xrow + kn + 256, 0, 3);
            }

            // B operands from LDS, 2-deep ping-pong: loads for tile nt+1
            // issue before the WMMA on tile nt, hiding the LDS latency.
            const __bf16* sBk = &sB[l * K_STAGE + kc + hlf * 8];
            BF16Vec bb[2];
            bb[0].h.lo = *reinterpret_cast<const bf16x8*>(sBk);
            bb[0].h.hi = *reinterpret_cast<const bf16x8*>(sBk + 16);
#pragma unroll
            for (int nt = 0; nt < 8; ++nt) {
                if (nt < 7) {
                    const __bf16* wr = sBk + (size_t)(nt + 1) * 16 * K_STAGE;
                    bb[(nt + 1) & 1].h.lo = *reinterpret_cast<const bf16x8*>(wr);
                    bb[(nt + 1) & 1].h.hi = *reinterpret_cast<const bf16x8*>(wr + 16);
                }
                acc[nt] = __builtin_amdgcn_wmma_f32_16x16x32_bf16(
                    false, a.v, false, bb[nt & 1].v, (short)0, acc[nt],
                    false, false);
            }
        }
        __syncthreads();   // sB consumed; safe to overwrite next stage
    }

    // Scatter D (M = r + 8*hlf, N = nt*16 + l) into time-major xw.
#pragma unroll
    for (int nt = 0; nt < 8; ++nt) {
#pragma unroll
        for (int r = 0; r < 8; ++r) {
            int m    = mtile * 16 + r + 8 * hlf;   // flat bt = b*SEQ + t
            int bidx = m >> 9;
            int tidx = m & (SEQ - 1);
            xw[((size_t)tidx * BATCH + bidx) * HIDDEN + nt * 16 + l] = acc[nt][r];
        }
    }
}

// ---------------------------------------------------------------------------
// Kernel 2: sequential recurrence h_t = tanh(xw_t + h_{t-1} W_hh^T), then
// logits = h_last . W_fc + b_fc.  Batch-parallel: 4 blocks of 16 rows each.
// W_hh (bf16) and h live in LDS; each step = 4 K-chunks of WMMA per tile.
// ---------------------------------------------------------------------------
__global__ __launch_bounds__(128) void rnn_recur(
        const float* __restrict__ xw,                // [SEQ][BATCH][HIDDEN]
        const float* __restrict__ W_hh,              // [HIDDEN][HIDDEN] fp32
        const float* __restrict__ W_fc,              // [1][HIDDEN]
        const float* __restrict__ b_fc,              // [1]
        float* __restrict__ out)                     // [BATCH]
{
    __shared__ __align__(16) __bf16 sW[HIDDEN * HIDDEN]; // W_hh bf16, [j][k]
    __shared__ __align__(16) __bf16 sH[16 * HIDDEN];     // current h bf16, [m][k]
    __shared__ float                sHf[16 * HIDDEN];    // fp32 h of last step

    const int tid  = threadIdx.x;
    const int wave = tid >> 5;
    const int lane = tid & 31;
    const int hlf  = lane >> 4;
    const int l    = lane & 15;
    const int b0   = blockIdx.x * 16;                // this block's batch rows
    const int nt0  = wave * 2;                       // two N-tiles per wave

    for (int i = tid; i < HIDDEN * HIDDEN; i += 128) sW[i] = (__bf16)W_hh[i];
    for (int i = tid; i < 16 * HIDDEN;     i += 128) sH[i] = (__bf16)0.0f;
    __syncthreads();

    for (int t = 0; t < SEQ; ++t) {
        // Prefetch the xw slab two steps ahead (8 KB / block).
        if (t + 2 < SEQ)
            __builtin_prefetch(
                xw + ((size_t)(t + 2) * BATCH + b0) * HIDDEN + tid * 16, 0, 3);

        // Accumulator init = xw_t tile (contiguous slab, coalesced dword loads).
        f32x8 acc[2];
#pragma unroll
        for (int j = 0; j < 2; ++j) {
            const float* cb =
                xw + ((size_t)t * BATCH + b0) * HIDDEN + (nt0 + j) * 16 + l;
#pragma unroll
            for (int r = 0; r < 8; ++r) acc[j][r] = cb[(r + 8 * hlf) * HIDDEN];
        }

        // h_{t-1} @ W_hh^T : K = 128 in four 32-chunks.
#pragma unroll
        for (int kc = 0; kc < 4; ++kc) {
            const int koff = kc * 32 + hlf * 8;
            BF16Vec a, bm[2];
            a.h.lo = *reinterpret_cast<const bf16x8*>(&sH[l * HIDDEN + koff]);
            a.h.hi = *reinterpret_cast<const bf16x8*>(&sH[l * HIDDEN + koff + 16]);
#pragma unroll
            for (int j = 0; j < 2; ++j) {
                const __bf16* wr = &sW[((nt0 + j) * 16 + l) * HIDDEN + koff];
                bm[j].h.lo = *reinterpret_cast<const bf16x8*>(wr);
                bm[j].h.hi = *reinterpret_cast<const bf16x8*>(wr + 16);
            }
#pragma unroll
            for (int j = 0; j < 2; ++j)
                acc[j] = __builtin_amdgcn_wmma_f32_16x16x32_bf16(
                    false, a.v, false, bm[j].v, (short)0, acc[j], false, false);
        }
        __syncthreads();   // all waves finished reading old h

        // tanh + writeback of h_t (each (m,n) owned by exactly one lane).
#pragma unroll
        for (int j = 0; j < 2; ++j) {
#pragma unroll
            for (int r = 0; r < 8; ++r) {
                float v = tanhf(acc[j][r]);
                int m = r + 8 * hlf;
                int n = (nt0 + j) * 16 + l;
                sH[m * HIDDEN + n] = (__bf16)v;
                if (t == SEQ - 1) sHf[m * HIDDEN + n] = v;
            }
        }
        __syncthreads();   // h_t visible before next step reads it
    }

    // Final projection: 16 rows per block.
    if (tid < 16) {
        float s = 0.f;
        for (int k = 0; k < HIDDEN; ++k) s += sHf[tid * HIDDEN + k] * W_fc[k];
        out[b0 + tid] = s + b_fc[0];
    }
}

// ---------------------------------------------------------------------------
extern "C" void kernel_launch(void* const* d_in, const int* in_sizes, int n_in,
                              void* d_out, int out_size, void* d_ws, size_t ws_size,
                              hipStream_t stream) {
    const float* x    = (const float*)d_in[0];   // [64,512,2048]
    const float* W_ih = (const float*)d_in[1];   // [128,2048]
    const float* W_hh = (const float*)d_in[2];   // [128,128]
    const float* b_ih = (const float*)d_in[3];   // [128]
    const float* b_hh = (const float*)d_in[4];   // [128]
    const float* W_fc = (const float*)d_in[5];   // [1,128]
    const float* b_fc = (const float*)d_in[6];   // [1]
    float* out = (float*)d_out;                  // [64,1] fp32

    // Workspace: [0, 512KB) = W_ih bf16; [512KB, 512KB+16MB) = xw fp32.
    __bf16* wih_bf = (__bf16*)d_ws;
    float* xw = (float*)((char*)d_ws +
                         (size_t)HIDDEN * INPUT_DIM * sizeof(__bf16));

    hipLaunchKernelGGL(cvt_to_bf16, dim3(256), dim3(256), 0, stream,
                       W_ih, wih_bf, HIDDEN * INPUT_DIM);

    // 32768 M-rows / (16 rows * 8 waves) = 256 blocks; 64 KB dynamic LDS.
    hipLaunchKernelGGL(xw_gemm, dim3(256), dim3(256),
                       HIDDEN * K_STAGE * sizeof(__bf16), stream,
                       x, wih_bf, b_ih, b_hh, xw);

    hipLaunchKernelGGL(rnn_recur, dim3(BATCH / 16), dim3(128), 0, stream,
                       xw, W_hh, W_fc, b_fc, out);
}
